// GCN_16922171146785
// MI455X (gfx1250) — matrix operand order repaired and checked by reading, not compiled, not run on hardware
//
#include <hip/hip_runtime.h>
#include <stdint.h>

// ---------------- constants from the reference ----------------
#define NN 30000          // nodes
#define NE 480000         // edges
#define NG 64             // graphs
#define EMB 448           // 2*128 + 96 + 96
#define QKV3 1344         // 3*EMB

typedef __attribute__((ext_vector_type(16))) _Float16 v16h;
typedef __attribute__((ext_vector_type(4)))  _Float16 v4h;
typedef __attribute__((ext_vector_type(8)))  float    v8f;

enum { ACT_NONE = 0, ACT_RELU = 1, ACT_LEAKY = 2 };

static inline unsigned cdivu(unsigned a, unsigned b) { return (a + b - 1u) / b; }

// Monotonic float<->int encoding for atomic max on floats.
__device__ inline int   fenc(float f) { int i = __float_as_int(f); return i >= 0 ? i : (i ^ 0x7FFFFFFF); }
__device__ inline float fdec(int i)   { return __int_as_float(i >= 0 ? i : (i ^ 0x7FFFFFFF)); }

// =====================================================================
// WMMA GEMM, fragment-ordered LDS staging.
//   C  = A @ B  (+bias)      [and if DUAL:  C2 = A @ B2 (+bias2)]
// A: row-major f32 [M,K] (leading dim lda). B: row-major [K,N] (ldb = N),
// or if BT: row-major [N,K] used as its transpose (ldb = K row stride).
// Block: 256 threads = 8 waves; block tile = 128 rows x (16*NSUB) cols.
// Wave w owns rows [16w,16w+16) of the block and does NSUB (x2 if DUAL)
// v_wmma_f32_16x16x32_f16 per 32-wide K step from one shared A fragment.
// Requires M%16==0, N%(16*NSUB)==0, K%32==0, 16B-aligned rows (true here).
// =====================================================================
template <int NSUB, bool BT, bool DUAL>
__launch_bounds__(256)
__global__ void gemm_wmma(const float* __restrict__ A, int lda,
                          const float* __restrict__ B, const float* __restrict__ B2, int ldb,
                          const float* __restrict__ bias, const float* __restrict__ bias2,
                          float* __restrict__ C, float* __restrict__ C2, int ldc,
                          int M, int N, int K)
{
    constexpr int BN = NSUB * 16;
    __shared__ alignas(32) _Float16 sA[8 * 512];                 // 8 row tiles, frag order
    __shared__ alignas(32) _Float16 sB[NSUB * 512];
    __shared__ alignas(32) _Float16 sB2[DUAL ? NSUB * 512 : 16];

    const int t      = threadIdx.x;
    const int lane   = t & 31;
    const int wv     = t >> 5;
    const int nb     = blockIdx.x * BN;
    const int mtiles = M >> 4;
    const int rowTile = blockIdx.y * 8 + wv;
    const bool active = rowTile < mtiles;
    const int mBlock  = blockIdx.y * 128;

    const v8f vzero = {0.f, 0.f, 0.f, 0.f, 0.f, 0.f, 0.f, 0.f};
    v8f acc[NSUB];
    v8f acc2[DUAL ? NSUB : 1];
#pragma unroll
    for (int c = 0; c < NSUB; ++c) acc[c] = vzero;
    if constexpr (DUAL) {
#pragma unroll
        for (int c = 0; c < NSUB; ++c) acc2[c] = vzero;
    }

    for (int kb = 0; kb < K; kb += 32) {
        __syncthreads();                       // fragments of previous step consumed
        // ---- stage A: 128x32 f32 -> f16 fragments (float4 loads, v4h stores)
#pragma unroll
        for (int j = 0; j < 4; ++j) {
            const int q   = t + j * 256;       // 1024 float4 slots
            const int row = q >> 3;            // 0..127
            const int c4  = (q & 7) << 2;      // 0,4,...,28 (kk base)
            int gr = mBlock + row; if (gr >= M) gr = M - 1;
            const float4 v = *(const float4*)(A + (size_t)gr * lda + (kb + c4));
            // A-frag: L = (m&15) + ((kk&16)?16:0), e = kk&15
            const int L  = (row & 15) + ((c4 & 16) ? 16 : 0);
            const int e0 = c4 & 15;
            v4h hv = { (_Float16)v.x, (_Float16)v.y, (_Float16)v.z, (_Float16)v.w };
            *(v4h*)&sA[(row >> 4) * 512 + L * 16 + e0] = hv;
        }
        // ---- stage B (and B2): 32 x BN
        constexpr int NB4 = (32 * BN) / 4 / 256;   // float4s per thread (1 or 2)
#pragma unroll
        for (int j = 0; j < NB4; ++j) {
            const int q = t + j * 256;
            if constexpr (!BT) {
                const int kk = q / (BN / 4);
                const int n4 = (q % (BN / 4)) << 2;
                const float4 v = *(const float4*)(B + (size_t)(kb + kk) * ldb + (nb + n4));
                // B-frag: L = (n&15) + ((kk&16)?16:0), e = kk&15
                const int ct = n4 >> 4;
                const int L0 = (n4 & 15) + ((kk & 16) ? 16 : 0);
                const int e  = kk & 15;
                _Float16* p = &sB[ct * 512 + L0 * 16 + e];
                p[0]  = (_Float16)v.x; p[16] = (_Float16)v.y;
                p[32] = (_Float16)v.z; p[48] = (_Float16)v.w;
            } else {
                const int nn = q >> 3;             // 0..BN-1
                const int k4 = (q & 7) << 2;       // 0..28
                const float4 v = *(const float4*)(B + (size_t)(nb + nn) * ldb + (kb + k4));
                const int ct = nn >> 4;
                const int L  = (nn & 15) + ((k4 & 16) ? 16 : 0);
                const int e0 = k4 & 15;
                v4h hv = { (_Float16)v.x, (_Float16)v.y, (_Float16)v.z, (_Float16)v.w };
                *(v4h*)&sB[ct * 512 + L * 16 + e0] = hv;
            }
        }
        if constexpr (DUAL) {
#pragma unroll
            for (int j = 0; j < NB4; ++j) {
                const int q = t + j * 256;
                const int kk = q / (BN / 4);
                const int n4 = (q % (BN / 4)) << 2;
                const float4 v = *(const float4*)(B2 + (size_t)(kb + kk) * ldb + (nb + n4));
                const int ct = n4 >> 4;
                const int L0 = (n4 & 15) + ((kk & 16) ? 16 : 0);
                const int e  = kk & 15;
                _Float16* p = &sB2[ct * 512 + L0 * 16 + e];
                p[0]  = (_Float16)v.x; p[16] = (_Float16)v.y;
                p[32] = (_Float16)v.z; p[48] = (_Float16)v.w;
            }
        }
        __syncthreads();
        // ---- compute: contiguous 32B fragment reads, NSUB(+NSUB) WMMAs
        const v16h af = *(const v16h*)&sA[wv * 512 + lane * 16];
#pragma unroll
        for (int c = 0; c < NSUB; ++c) {
            const v16h bf = *(const v16h*)&sB[c * 512 + lane * 16];
            acc[c] = __builtin_amdgcn_wmma_f32_16x16x32_f16(false, af, false, bf,
                                                            (short)0, acc[c], false, false);
        }
        if constexpr (DUAL) {
#pragma unroll
            for (int c = 0; c < NSUB; ++c) {
                const v16h bf = *(const v16h*)&sB2[c * 512 + lane * 16];
                acc2[c] = __builtin_amdgcn_wmma_f32_16x16x32_f16(false, af, false, bf,
                                                                 (short)0, acc2[c], false, false);
            }
        }
    }

    if (!active) return;
    const int mb = rowTile << 4;
    const int rb = (lane & 16) ? 8 : 0;     // C layout: lanes 16-31 hold M=8..15
    const int c0 = lane & 15;
#pragma unroll
    for (int c = 0; c < NSUB; ++c) {
        const int col = nb + c * 16 + c0;
        const float bv = bias ? bias[col] : 0.f;
#pragma unroll
        for (int r = 0; r < 8; ++r)
            C[(size_t)(mb + rb + r) * ldc + col] = acc[c][r] + bv;
    }
    if constexpr (DUAL) {
#pragma unroll
        for (int c = 0; c < NSUB; ++c) {
            const int col = nb + c * 16 + c0;
            const float bv = bias2 ? bias2[col] : 0.f;
#pragma unroll
            for (int r = 0; r < 8; ++r)
                C2[(size_t)(mb + rb + r) * ldc + col] = acc2[c][r] + bv;
        }
    }
}

// Dual-output GEMM: C = A@B (+bias), C2 = A@B2 (+bias2); B row-major [K,N].
static void gemm_dual(hipStream_t st, const float* A, int lda,
                      const float* B, const float* B2,
                      const float* bias, const float* bias2,
                      float* C, float* C2, int M, int N, int K)
{
    const unsigned mt = cdivu((unsigned)(M / 16), 8u);
    if (N % 64 == 0) {
        dim3 g(N / 64, mt);
        gemm_wmma<4, false, true><<<g, 256, 0, st>>>(A, lda, B, B2, N, bias, bias2, C, C2, N, M, N, K);
    } else {
        dim3 g(N / 32, mt);
        gemm_wmma<2, false, true><<<g, 256, 0, st>>>(A, lda, B, B2, N, bias, bias2, C, C2, N, M, N, K);
    }
}

// Single-output GEMM with transposed B: C = A@Bt + bias; B row-major [N,K].
static void gemm_bt(hipStream_t st, const float* A, int lda,
                    const float* B, int ldb, const float* bias,
                    float* C, int M, int N, int K)
{
    const unsigned mt = cdivu((unsigned)(M / 16), 8u);
    dim3 g(N / 64, mt);
    gemm_wmma<4, true, false><<<g, 256, 0, st>>>(A, lda, B, nullptr, ldb, bias, nullptr, C, nullptr, N, M, N, K);
}

// =====================================================================
// Graph primitives
// =====================================================================
__global__ void k_count(const int* __restrict__ idx, int n, int* __restrict__ cnt) {
    int i = blockIdx.x * blockDim.x + threadIdx.x;
    if (i < n) atomicAdd(&cnt[idx[i]], 1);
}

__global__ void k_inv(const int* __restrict__ cnt, float* __restrict__ inv, int n) {
    int i = blockIdx.x * blockDim.x + threadIdx.x;
    if (i < n) inv[i] = 1.f / (float)max(cnt[i], 1);
}

__global__ void k_init_i32(int* __restrict__ p, int v, int n) {
    int i = blockIdx.x * blockDim.x + threadIdx.x;
    if (i < n) p[i] = v;
}

// out[d[e]*F + f4..f4+3] += feat[s[e]*F + f4..f4+3]
__global__ void k_scatter_add4(const float* __restrict__ feat, const int* __restrict__ sA_,
                               const int* __restrict__ dA_, int E, int F, float* __restrict__ out) {
    unsigned i = blockIdx.x * blockDim.x + threadIdx.x;
    const unsigned F4 = (unsigned)F >> 2;
    if (i >= (unsigned)E * F4) return;
    unsigned e = i / F4, f4 = (i - e * F4) << 2;
    const int s = sA_[e], d = dA_[e];
    const float4 v = *(const float4*)&feat[(size_t)s * F + f4];
    float* o = &out[(size_t)d * F + f4];
    atomicAdd(o + 0, v.x); atomicAdd(o + 1, v.y);
    atomicAdd(o + 2, v.z); atomicAdd(o + 3, v.w);
}

// SAGE epilogue (in place): uo = act( uo + bl[col] + inv[row]*agg )
__global__ void k_sage_ep(float* __restrict__ uo, const float* __restrict__ agg,
                          const float* __restrict__ inv, const float* __restrict__ bl,
                          int F, int act, float slope) {
    unsigned i = blockIdx.x * blockDim.x + threadIdx.x;
    if (i >= (unsigned)NN * (unsigned)F) return;
    unsigned row = i / F, col = i - row * F;
    float v = uo[i] + bl[col] + inv[row] * agg[i];
    if (act == ACT_RELU)       v = fmaxf(v, 0.f);
    else if (act == ACT_LEAKY) v = v > 0.f ? v : slope * v;
    uo[i] = v;
}

__global__ void k_rows_bias(float* __restrict__ out, const float* __restrict__ bias, int n, int F) {
    unsigned i = blockIdx.x * blockDim.x + threadIdx.x;
    if (i < (unsigned)n * (unsigned)F) out[i] = bias[i % F];
}

// ---------------- GATv2 (self loops = edge ids >= E) ----------------
__global__ void k_gat_logits(const float* __restrict__ xl, const float* __restrict__ xr,
                             const float* __restrict__ att, const int* __restrict__ sA_,
                             const int* __restrict__ dA_, int E, int Nn, int H, int C,
                             float* __restrict__ elog, int* __restrict__ menc) {
    int i = blockIdx.x * blockDim.x + threadIdx.x;
    int tot = (E + Nn) * H;
    if (i >= tot) return;
    int e = i / H, h = i - e * H;
    int s = e < E ? sA_[e] : e - E;
    int d = e < E ? dA_[e] : e - E;
    const float* pl = xl + (size_t)s * H * C + h * C;
    const float* pr = xr + (size_t)d * H * C + h * C;
    const float* pa = att + h * C;
    float acc = 0.f;
    for (int c = 0; c < C; ++c) {
        float tt = pl[c] + pr[c];
        tt = tt > 0.f ? tt : 0.2f * tt;
        acc += tt * pa[c];
    }
    elog[i] = acc;
    atomicMax(menc + (size_t)d * H + h, fenc(acc));
}

__global__ void k_gat_expz(float* __restrict__ elog, const int* __restrict__ menc,
                           const int* __restrict__ dA_, int E, int Nn, int H,
                           float* __restrict__ z) {
    int i = blockIdx.x * blockDim.x + threadIdx.x;
    int tot = (E + Nn) * H;
    if (i >= tot) return;
    int e = i / H, h = i - e * H;
    int d = e < E ? dA_[e] : e - E;
    float m = fdec(menc[(size_t)d * H + h]);
    float w = __expf(elog[i] - m);
    elog[i] = w;
    atomicAdd(&z[(size_t)d * H + h], w);
}

__global__ void k_gat_coef(float* __restrict__ elog, const float* __restrict__ z,
                           const int* __restrict__ dA_, int E, int Nn, int H) {
    int i = blockIdx.x * blockDim.x + threadIdx.x;
    int tot = (E + Nn) * H;
    if (i >= tot) return;
    int e = i / H, h = i - e * H;
    int d = e < E ? dA_[e] : e - E;
    elog[i] /= z[(size_t)d * H + h];
}

__global__ void k_gat_scatter4(const float* __restrict__ xl, const float* __restrict__ elog,
                               const int* __restrict__ sA_, const int* __restrict__ dA_,
                               int E, int Nn, int H, int C, float* __restrict__ out) {
    unsigned i = blockIdx.x * blockDim.x + threadIdx.x;
    const unsigned HC4 = (unsigned)(H * C) >> 2;
    if (i >= (unsigned)(E + Nn) * HC4) return;
    unsigned e = i / HC4, r4 = (i - e * HC4) << 2;
    int s = (int)e < E ? sA_[e] : (int)e - E;
    int d = (int)e < E ? dA_[e] : (int)e - E;
    const float coef = elog[(size_t)e * H + (int)r4 / C];   // C%4==0 so r4..r4+3 same head
    const float4 v = *(const float4*)&xl[(size_t)s * H * C + r4];
    float* o = &out[(size_t)d * H * C + r4];
    atomicAdd(o + 0, v.x * coef); atomicAdd(o + 1, v.y * coef);
    atomicAdd(o + 2, v.z * coef); atomicAdd(o + 3, v.w * coef);
}

// ---------------- pooling + head ----------------
__global__ void k_pool(const float* __restrict__ x1, const float* __restrict__ x2,
                       const float* __restrict__ x3, const int* __restrict__ batch,
                       int Nn, int* __restrict__ genc) {
    unsigned i = blockIdx.x * blockDim.x + threadIdx.x;
    if (i >= (unsigned)Nn * EMB) return;
    unsigned n = i / EMB, c = i - n * EMB;
    float v;
    if (c < 128u)       v = x1[(size_t)n * 128 + c];
    else if (c < 224u)  v = x3[(size_t)n * 96 + (c - 128u)];
    else if (c < 352u)  v = x2[(size_t)n * 128 + (c - 224u)];
    else                v = x3[(size_t)n * 96 + (c - 352u)];
    atomicMax(genc + (size_t)batch[n] * EMB + c, fenc(v));
}

__global__ void k_dec(const int* __restrict__ genc, float* __restrict__ g, int n) {
    int i = blockIdx.x * blockDim.x + threadIdx.x;
    if (i < n) g[i] = fdec(genc[i]);
}

// 8 heads, head dim 56, L = 64; one block per head, thread = query row.
__global__ void k_mha(const float* __restrict__ qkv, float* __restrict__ o) {
    const int h = blockIdx.x;
    const int t = threadIdx.x;
    __shared__ float kb[64][56];
    __shared__ float vb[64][56];
    for (int c = 0; c < 56; ++c) {
        kb[t][c] = qkv[t * QKV3 + EMB     + h * 56 + c];
        vb[t][c] = qkv[t * QKV3 + 2 * EMB + h * 56 + c];
    }
    __syncthreads();
    float q[56];
    for (int c = 0; c < 56; ++c) q[c] = qkv[t * QKV3 + h * 56 + c];
    const float scale = 0.13363062095621219f;  // 1/sqrt(56)
    float m = -3.0e38f, s = 0.f, oa[56];
    for (int c = 0; c < 56; ++c) oa[c] = 0.f;
    for (int j = 0; j < 64; ++j) {
        float dot = 0.f;
        for (int c = 0; c < 56; ++c) dot += q[c] * kb[j][c];
        dot *= scale;
        float mn = fmaxf(m, dot);
        float corr = __expf(m - mn);
        float w = __expf(dot - mn);
        s = s * corr + w;
        for (int c = 0; c < 56; ++c) oa[c] = oa[c] * corr + w * vb[j][c];
        m = mn;
    }
    float inv = 1.f / s;
    for (int c = 0; c < 56; ++c) o[t * EMB + h * 56 + c] = oa[c] * inv;
}

__global__ void k_final(const float* __restrict__ g2, const float* __restrict__ W,
                        const float* __restrict__ b, float* __restrict__ out) {
    int i = threadIdx.x;
    if (i >= NG * 3) return;
    int gi = i / 3, j = i - gi * 3;
    float acc = b[j];
    for (int k = 0; k < EMB; ++k) acc += g2[(size_t)gi * EMB + k] * W[k * 3 + j];
    out[i] = acc;
}

// =====================================================================
// host-side building blocks
// =====================================================================
// SAGE layer, transform-before-aggregate with fused dual GEMM:
//   {t, u} = {x@Wl, x@Wr} ; agg = scatter_add(t, s->d) ;
//   out(=u in place) = act(u + bl + inv[row]*agg)
static void sage_layer(hipStream_t st, const float* xin, int lda, int K,
                       const float* Wl, const float* bl, const float* Wr,
                       const int* sArr, const int* dArr, const float* inv,
                       float* tbuf, float* agg, float* out, int Nout, int act, float slope)
{
    gemm_dual(st, xin, lda, Wl, Wr, nullptr, nullptr, tbuf, out, NN, Nout, K);
    hipMemsetAsync(agg, 0, (size_t)NN * Nout * sizeof(float), st);
    unsigned tot = (unsigned)NE * (unsigned)(Nout / 4);
    k_scatter_add4<<<cdivu(tot, 256), 256, 0, st>>>(tbuf, sArr, dArr, NE, Nout, agg);
    unsigned tot2 = (unsigned)NN * (unsigned)Nout;
    k_sage_ep<<<cdivu(tot2, 256), 256, 0, st>>>(out, agg, inv, bl, Nout, act, slope);
}

static void gat_run(hipStream_t st, const float* xin, int lda, int K,
                    const float* Wl, const float* bl, const float* Wr, const float* br,
                    const float* att, const float* bias,
                    const int* sArr, const int* dArr, int H, int C,
                    float* xl, float* xr, float* elog, int* menc, float* z, float* out)
{
    const int HC = H * C;
    gemm_dual(st, xin, lda, Wl, Wr, bl, br, xl, xr, NN, HC, K);
    k_init_i32<<<cdivu(NN * H, 256), 256, 0, st>>>(menc, (int)0x80000000, NN * H);
    hipMemsetAsync(z, 0, (size_t)NN * H * sizeof(float), st);
    const int EN = NE + NN;
    k_gat_logits <<<cdivu(EN * H, 128), 128, 0, st>>>(xl, xr, att, sArr, dArr, NE, NN, H, C, elog, menc);
    k_gat_expz   <<<cdivu(EN * H, 256), 256, 0, st>>>(elog, menc, dArr, NE, NN, H, z);
    k_gat_coef   <<<cdivu(EN * H, 256), 256, 0, st>>>(elog, z, dArr, NE, NN, H);
    k_rows_bias  <<<cdivu(NN * HC, 256), 256, 0, st>>>(out, bias, NN, HC);
    unsigned tot = (unsigned)EN * (unsigned)(HC / 4);
    k_gat_scatter4<<<cdivu(tot, 256), 256, 0, st>>>(xl, elog, sArr, dArr, NE, NN, H, C, out);
}

// =====================================================================
extern "C" void kernel_launch(void* const* d_in, const int* in_sizes, int n_in,
                              void* d_out, int out_size, void* d_ws, size_t ws_size,
                              hipStream_t stream)
{
    (void)in_sizes; (void)out_size;
    if (n_in < 52) return;

    const float* x     = (const float*)d_in[0];        // [NN,1536]
    const int*   ei    = (const int*)d_in[1];          // [2,NE]
    const int*   src   = ei;
    const int*   dst   = ei + NE;
    const int*   batch = (const int*)d_in[2];

    int p = 4;
    const float *c1pWl=(const float*)d_in[p++], *c1pbl=(const float*)d_in[p++], *c1pWr=(const float*)d_in[p++];
    const float *c2pWl=(const float*)d_in[p++], *c2pbl=(const float*)d_in[p++], *c2pWr=(const float*)d_in[p++];
    const float *c3pWl=(const float*)d_in[p++], *c3pbl=(const float*)d_in[p++], *c3pWr=(const float*)d_in[p++];
    const float *c1dWl=(const float*)d_in[p++], *c1dbl=(const float*)d_in[p++], *c1dWr=(const float*)d_in[p++];
    const float *c2dWl=(const float*)d_in[p++], *c2dbl=(const float*)d_in[p++], *c2dWr=(const float*)d_in[p++];
    const float *c3dWl=(const float*)d_in[p++], *c3dbl=(const float*)d_in[p++], *c3dWr=(const float*)d_in[p++];
    const float *c1sWl=(const float*)d_in[p++], *c1sbl=(const float*)d_in[p++], *c1sWr=(const float*)d_in[p++];
    const float *c2sWl=(const float*)d_in[p++], *c2sbl=(const float*)d_in[p++], *c2sWr=(const float*)d_in[p++];
    const float *apWl =(const float*)d_in[p++], *apbl =(const float*)d_in[p++], *apWr =(const float*)d_in[p++];
    const float *apbr =(const float*)d_in[p++], *apatt=(const float*)d_in[p++], *apbias=(const float*)d_in[p++];
    const float *adWl =(const float*)d_in[p++], *adbl =(const float*)d_in[p++], *adWr =(const float*)d_in[p++];
    const float *adbr =(const float*)d_in[p++], *adatt=(const float*)d_in[p++], *adbias=(const float*)d_in[p++];
    const float *asWl =(const float*)d_in[p++], *asbl =(const float*)d_in[p++], *asWr =(const float*)d_in[p++];
    const float *asbr =(const float*)d_in[p++], *asatt=(const float*)d_in[p++], *asbias=(const float*)d_in[p++];
    const float *mWin =(const float*)d_in[p++], *mbin =(const float*)d_in[p++];
    const float *mWout=(const float*)d_in[p++], *mbout=(const float*)d_in[p++];
    const float *linW =(const float*)d_in[p++], *linb =(const float*)d_in[p++];

    // ---------------- workspace layout ----------------
    uint8_t* wsp = (uint8_t*)d_ws;
    size_t off = 0;
    auto take = [&](size_t bytes) -> void* {
        void* r = wsp + off;
        off += (bytes + 255) & ~(size_t)255;
        return r;
    };
    int*   cnt_f = (int*)  take((size_t)NN * 4);
    int*   cnt_r = (int*)  take((size_t)NN * 4);
    float* inv_f = (float*)take((size_t)NN * 4);
    float* inv_r = (float*)take((size_t)NN * 4);
    float* bufT  = (float*)take((size_t)NN * 128 * 4);
    float* bufAg = (float*)take((size_t)NN * 128 * 4);
    float* bufU  = (float*)take((size_t)NN * 128 * 4);
    float* bufV  = (float*)take((size_t)NN * 128 * 4);
    float* x1f   = (float*)take((size_t)NN * 128 * 4);
    float* x2f   = (float*)take((size_t)NN * 128 * 4);
    float* x3f   = (float*)take((size_t)NN * 96 * 4);
    float* elog  = (float*)take((size_t)(NE + NN) * 3 * 4);
    int*   menc  = (int*)  take((size_t)NN * 3 * 4);
    float* zbuf  = (float*)take((size_t)NN * 3 * 4);
    int*   genc  = (int*)  take((size_t)NG * EMB * 4);
    float* gbuf  = (float*)take((size_t)NG * EMB * 4);
    float* qkv   = (float*)take((size_t)NG * QKV3 * 4);
    float* attn  = (float*)take((size_t)NG * EMB * 4);
    float* g2b   = (float*)take((size_t)NG * EMB * 4);
    if (off > ws_size) return;

    // ---------------- degree counts (both directions) ----------------
    hipMemsetAsync(cnt_f, 0, (size_t)NN * 4, stream);
    hipMemsetAsync(cnt_r, 0, (size_t)NN * 4, stream);
    k_count<<<cdivu(NE, 256), 256, 0, stream>>>(dst, NE, cnt_f);
    k_count<<<cdivu(NE, 256), 256, 0, stream>>>(src, NE, cnt_r);
    k_inv<<<cdivu(NN, 256), 256, 0, stream>>>(cnt_f, inv_f, NN);
    k_inv<<<cdivu(NN, 256), 256, 0, stream>>>(cnt_r, inv_r, NN);

    // ---------------- branch p (forward edges) ----------------
    sage_layer(stream, x,       1536, 384, c1pWl, c1pbl, c1pWr, src, dst, inv_f, bufT, bufAg, bufU, 128, ACT_RELU, 0.f);
    sage_layer(stream, bufU,    128,  128, c2pWl, c2pbl, c2pWr, src, dst, inv_f, bufT, bufAg, bufV, 128, ACT_RELU, 0.f);
    sage_layer(stream, bufV,    128,  128, c3pWl, c3pbl, c3pWr, src, dst, inv_f, bufT, bufAg, bufU, 128, ACT_NONE, 0.f);
    gat_run(stream, bufU, 128, 128, apWl, apbl, apWr, apbr, apatt, apbias,
            src, dst, 1, 128, bufT, bufAg, elog, menc, zbuf, x1f);

    // ---------------- branch d (reversed edges) ----------------
    sage_layer(stream, x + 384, 1536, 384, c1dWl, c1dbl, c1dWr, dst, src, inv_r, bufT, bufAg, bufU, 128, ACT_RELU, 0.f);
    sage_layer(stream, bufU,    128,  128, c2dWl, c2dbl, c2dWr, dst, src, inv_r, bufT, bufAg, bufV, 128, ACT_RELU, 0.f);
    sage_layer(stream, bufV,    128,  128, c3dWl, c3dbl, c3dWr, dst, src, inv_r, bufT, bufAg, bufU, 128, ACT_NONE, 0.f);
    gat_run(stream, bufU, 128, 128, adWl, adbl, adWr, adbr, adatt, adbias,
            dst, src, 1, 128, bufT, bufAg, elog, menc, zbuf, x2f);

    // ---------------- branch s (forward edges, LeakyReLU 0.1) ----------------
    sage_layer(stream, x + 768, 1536, 768, c1sWl, c1sbl, c1sWr, src, dst, inv_f, bufT, bufAg, bufU, 32, ACT_LEAKY, 0.1f);
    sage_layer(stream, bufU,    32,   32,  c2sWl, c2sbl, c2sWr, src, dst, inv_f, bufT, bufAg, bufV, 32, ACT_NONE, 0.f);
    gat_run(stream, bufV, 32, 32, asWl, asbl, asWr, asbr, asatt, asbias,
            src, dst, 3, 32, bufT, bufAg, elog, menc, zbuf, x3f);

    // ---------------- pooling: g = [segmax(x1|x3) | segmax(x2|x3)] ----------------
    k_init_i32<<<cdivu(NG * EMB, 256), 256, 0, stream>>>(genc, (int)0x80000000, NG * EMB);
    k_pool<<<cdivu((unsigned)NN * EMB, 256), 256, 0, stream>>>(x1f, x2f, x3f, batch, NN, genc);
    k_dec<<<cdivu(NG * EMB, 256), 256, 0, stream>>>(genc, gbuf, NG * EMB);

    // ---------------- MHA head + final linear ----------------
    gemm_bt(stream, gbuf, EMB, mWin, EMB, mbin, qkv, NG, QKV3, EMB);
    k_mha<<<8, 64, 0, stream>>>(qkv, attn);
    gemm_bt(stream, attn, EMB, mWout, EMB, mbout, g2b, NG, EMB, EMB);
    k_final<<<1, 192, 0, stream>>>(g2b, linW, linb, (float*)d_out);
}
